// DynamicLinkPredictor_42691974922346
// MI455X (gfx1250) — matrix operand order repaired and checked by reading, not compiled, not run on hardware
//
#include <hip/hip_runtime.h>
#include <hip/hip_bf16.h>
#include <math.h>

typedef __attribute__((ext_vector_type(16))) _Float16 v16h;
typedef __attribute__((ext_vector_type(8)))  _Float16 v8h;
typedef __attribute__((ext_vector_type(8)))  float    v8f;

#define NN    768
#define HIDD  128
#define TWOH  256
#define EPSF  1e-5f
#define NTOTF 589824.0f

// ---------------------------------------------------------------------------
// Workspace layout (float offsets unless noted)
// ---------------------------------------------------------------------------
#define O_H      0          // 768*128
#define O_HN     98304      // 768*128
#define O_MU     196608     // 128
#define O_VAR    196736     // 128
#define O_A      196864     // 768*256
#define O_B      393472     // 768*256
#define O_ACC    590080     // 768 floats: bn1sum(256), bn1sq(256), bn2sum(128), bn2sq(128)
#define O_SCALE1 590848     // 256
#define O_SHIFT1 591104     // 256
#define O_C2P    591360     // 128
#define O_WP     591488     // 128
#define O_CONST  591616     // 16
#define O_W2HT_BYTES 2366528 // = 4*591632, 32B aligned; 32768 halfs (64KB)

// ---------------------------------------------------------------------------
// Zero the accumulator region (bn1/bn2 sums) each launch.
// ---------------------------------------------------------------------------
__global__ void zero_accum_kernel(float* acc) {
    if (threadIdx.x < 768) acc[threadIdx.x] = 0.0f;
}

// ---------------------------------------------------------------------------
// GCLSTM layer with H0=C0=0:
//   ig = sigmoid(h@W_i + conv_i_b + b_i)
//   cc = tanh  (h@W_c + conv_c_b + b_c)
//   C1 = ig*cc
//   og = sigmoid(h@W_o + conv_o_b + w_c_o*C1 + b_o)
//   hn = og*tanh(C1)
// grid = 768 (node), block = 128 (column)
// ---------------------------------------------------------------------------
__global__ __launch_bounds__(128) void gates_kernel(
    const float* __restrict__ h, int fin,
    const float* __restrict__ Wi, const float* __restrict__ Wc,
    const float* __restrict__ Wo,
    const float* __restrict__ bi, const float* __restrict__ bc,
    const float* __restrict__ bo,
    const float* __restrict__ cib, const float* __restrict__ ccb,
    const float* __restrict__ cob,
    const float* __restrict__ wco,
    float* __restrict__ hn)
{
    __shared__ float hrow[128];
    int n = blockIdx.x, c = threadIdx.x;
    if (c < fin) hrow[c] = h[n * fin + c];
    __syncthreads();
    float zi = 0.f, zc = 0.f, zo = 0.f;
    for (int k = 0; k < fin; ++k) {
        float hv = hrow[k];
        zi = fmaf(hv, Wi[k * HIDD + c], zi);
        zc = fmaf(hv, Wc[k * HIDD + c], zc);
        zo = fmaf(hv, Wo[k * HIDD + c], zo);
    }
    zi += cib[c] + bi[c];
    zc += ccb[c] + bc[c];
    float ig = 1.f / (1.f + expf(-zi));
    float cc = tanhf(zc);
    float C1 = ig * cc;
    zo += cob[c] + wco[c] * C1 + bo[c];
    float og = 1.f / (1.f + expf(-zo));
    hn[n * HIDD + c] = og * tanhf(C1);
}

// BatchNorm stats over the 768 nodes. grid = 128 (col), block = 256
__global__ __launch_bounds__(256) void bn_stats_kernel(
    const float* __restrict__ hn, float* __restrict__ mu, float* __restrict__ var)
{
    int c = blockIdx.x, t = threadIdx.x;
    float s = 0.f, q = 0.f;
    for (int n = t; n < NN; n += 256) {
        float v = hn[n * HIDD + c];
        s += v; q += v * v;
    }
    __shared__ float ss[256], qq[256];
    ss[t] = s; qq[t] = q;
    __syncthreads();
    for (int off = 128; off > 0; off >>= 1) {
        if (t < off) { ss[t] += ss[t + off]; qq[t] += qq[t + off]; }
        __syncthreads();
    }
    if (t == 0) {
        float m = ss[0] / (float)NN;
        mu[c] = m;
        var[c] = qq[0] / (float)NN - m * m;
    }
}

// Apply BN (+ optional residual). grid = 768, block = 128
__global__ __launch_bounds__(128) void bn_apply_kernel(
    const float* __restrict__ hn, const float* __restrict__ mu,
    const float* __restrict__ var, const float* __restrict__ g,
    const float* __restrict__ b, float* __restrict__ h, int residual)
{
    int n = blockIdx.x, c = threadIdx.x;
    float v = (hn[n * HIDD + c] - mu[c]) * rsqrtf(var[c] + EPSF) * g[c] + b[c];
    h[n * HIDD + c] = residual ? (h[n * HIDD + c] + v) : v;
}

// A = h @ W1[:128] + b1 ; B = h @ W1[128:]. grid = 768, block = 256
__global__ __launch_bounds__(256) void ab_kernel(
    const float* __restrict__ h, const float* __restrict__ W1,
    const float* __restrict__ b1, float* __restrict__ A, float* __restrict__ B)
{
    __shared__ float hrow[128];
    int n = blockIdx.x, t = threadIdx.x;
    if (t < 128) hrow[t] = h[n * HIDD + t];
    __syncthreads();
    float a = b1[t], bb = 0.f;
    for (int k = 0; k < 128; ++k) {
        float hv = hrow[k];
        a  = fmaf(hv, W1[k * TWOH + t], a);
        bb = fmaf(hv, W1[(128 + k) * TWOH + t], bb);
    }
    A[n * TWOH + t] = a;
    B[n * TWOH + t] = bb;
}

// BN1 stats over all 589824 pairs: relu(A[i,c]+B[j,c]). grid = 768 (i), block = 256 (c)
__global__ __launch_bounds__(256) void bn1_stats_kernel(
    const float* __restrict__ A, const float* __restrict__ B,
    float* __restrict__ bn1sum, float* __restrict__ bn1sq)
{
    int i = blockIdx.x, c = threadIdx.x;
    float a = A[i * TWOH + c];
    float s = 0.f, q = 0.f;
    for (int j = 0; j < NN; ++j) {
        float v = fmaxf(a + B[j * TWOH + c], 0.f);
        s += v; q = fmaf(v, v, q);
    }
    atomicAdd(&bn1sum[c], s);
    atomicAdd(&bn1sq[c], q);
}

// Fold BN1 into W2: scale1/shift1, c2p = b2 + shift1@W2. single block 256
__global__ __launch_bounds__(256) void prep1_kernel(
    const float* __restrict__ bn1sum, const float* __restrict__ bn1sq,
    const float* __restrict__ g1, const float* __restrict__ bb1,
    const float* __restrict__ b2, const float* __restrict__ W2,
    float* __restrict__ scale1, float* __restrict__ shift1, float* __restrict__ c2p)
{
    __shared__ float sh_shift[256];
    int t = threadIdx.x;
    float m  = bn1sum[t] / NTOTF;
    float v  = bn1sq[t] / NTOTF - m * m;
    float sc = g1[t] * rsqrtf(v + EPSF);
    float sf = bb1[t] - m * sc;
    scale1[t] = sc; shift1[t] = sf; sh_shift[t] = sf;
    __syncthreads();
    if (t < 128) {
        float acc = b2[t];
        for (int k = 0; k < 256; ++k) acc = fmaf(sh_shift[k], W2[k * HIDD + t], acc);
        c2p[t] = acc;
    }
}

// W2' transposed + f16: W2hT[c*256+k] = (half)(scale1[k]*W2[k,c]). grid=128 (c), block=256 (k)
__global__ __launch_bounds__(256) void w2ht_kernel(
    const float* __restrict__ scale1, const float* __restrict__ W2,
    _Float16* __restrict__ W2hT)
{
    int c = blockIdx.x, k = threadIdx.x;
    W2hT[c * 256 + k] = (_Float16)(scale1[k] * W2[k * HIDD + c]);
}

// Fold BN2+W3: wp[c] = g2*rs2*W3[c]; const = b3 + sum((bb2 - mu2*g2*rs2)*W3). single block 128
__global__ __launch_bounds__(128) void prep2_kernel(
    const float* __restrict__ bn2sum, const float* __restrict__ bn2sq,
    const float* __restrict__ g2, const float* __restrict__ bb2,
    const float* __restrict__ W3, const float* __restrict__ b3,
    float* __restrict__ wp, float* __restrict__ constv)
{
    __shared__ float red[128];
    int t = threadIdx.x;
    float m  = bn2sum[t] / NTOTF;
    float v  = bn2sq[t] / NTOTF - m * m;
    float sc = g2[t] * rsqrtf(v + EPSF);
    float w3 = W3[t];
    wp[t] = sc * w3;
    red[t] = (bb2[t] - m * sc) * w3;
    __syncthreads();
    for (int off = 64; off > 0; off >>= 1) {
        if (t < off) red[t] += red[t + off];
        __syncthreads();
    }
    if (t == 0) constv[0] = b3[0] + red[0];
}

// ---------------------------------------------------------------------------
// Heavy WMMA pass. grid = 2304 = 768 i * 3 j-blocks, block = 256 (8 waves).
// PASS 0: accumulate BN2 stats of z2 = relu(z1 @ W2' + c2p).
// PASS 1: out[i,j] = sigmoid(z2 . wp + const).
// ---------------------------------------------------------------------------
template <int PASS>
__global__ __launch_bounds__(256) void pair_head_kernel(
    const float* __restrict__ A, const float* __restrict__ Bm,
    const _Float16* __restrict__ W2hT, const float* __restrict__ c2p,
    float* __restrict__ bn2sum, float* __restrict__ bn2sq,
    const float* __restrict__ wp, const float* __restrict__ constv,
    float* __restrict__ outp)
{
    __shared__ __align__(32) float Ash[256];
    __shared__ __align__(32) char  tileRaw[8192]; // z1 f16 16x256  /  z2 f32 16x128
    __shared__ float wsum[128], wsq[128];         // pass 0
    __shared__ float wp_sh[128];                  // pass 1
    __shared__ float rowacc[16];                  // pass 1

    const int wg   = blockIdx.x;
    const int i    = wg / 3;
    const int jb   = wg % 3;
    const int tid  = threadIdx.x;
    const int lane = tid & 31;
    const int wv   = tid >> 5;               // wave id 0..7 -> output cols 16*wv..
    const int col  = wv * 16 + (lane & 15);  // this lane's output column (0..127)
    const int hi   = lane >> 4;              // 0 for lanes 0-15, 1 for 16-31

    Ash[tid] = A[i * TWOH + tid];
    if (PASS == 0) {
        if (tid < 128) { wsum[tid] = 0.f; wsq[tid] = 0.f; }
    } else {
        if (tid < 128) wp_sh[tid] = wp[tid];
    }
    __syncthreads();

    // Preload B-fragments (contiguous K order used here, same as A): 16 contiguous K halfs,
    // K-base = 16*hi within each 32-wide K window, column = lane&15.
    v16h bfr[8];
    const _Float16* w2c = W2hT + col * 256 + hi * 16;
#pragma unroll
    for (int s = 0; s < 8; ++s)
        bfr[s] = *(const v16h*)(w2c + s * 32);

    const float c2pv = c2p[col];
    const float cst  = (PASS == 1) ? constv[0] : 0.f;
    float accS = 0.f, accQ = 0.f;

    _Float16* z1h  = (_Float16*)tileRaw;
    float*    z2sh = (float*)tileRaw;

    for (int m = 0; m < 16; ++m) {
        const int j0 = jb * 256 + m * 16;
        // Build f16 z1 tile: 16 rows (pairs) x 256 K. Thread tid = column.
        {
            const float a = Ash[tid];
#pragma unroll
            for (int r = 0; r < 16; ++r) {
                float v = fmaxf(a + Bm[(j0 + r) * TWOH + tid], 0.f);
                z1h[r * 256 + tid] = (_Float16)v;
            }
        }
        __syncthreads();

        // 8-step K loop of WMMA. A operand in contiguous K order: lane row = lane&15,
        // per 32-K window: halfs [16*hi, 16*hi+16) (matches B fragments).
        v8f cf = {};
        const _Float16* rowp = z1h + (lane & 15) * 256 + hi * 16;
#pragma unroll
        for (int s = 0; s < 8; ++s) {
            union { v16h v; v8h h[2]; } au;
            au.h[0] = *(const v8h*)(rowp + s * 32);
            au.h[1] = *(const v8h*)(rowp + s * 32 + 8);
            cf = __builtin_amdgcn_wmma_f32_16x16x32_f16(
                false, au.v, false, bfr[s], (short)0, cf, false, false);
        }
        __syncthreads(); // everyone done reading z1h

        if (PASS == 0) {
#pragma unroll
            for (int k = 0; k < 8; ++k) {
                float z2 = fmaxf(cf[k] + c2pv, 0.f);
                accS += z2;
                accQ = fmaf(z2, z2, accQ);
            }
        } else {
            // C layout: VGPR k -> row k + 8*hi, col = lane&15 (+16*wv).
            if (tid < 16) rowacc[tid] = 0.f;
#pragma unroll
            for (int k = 0; k < 8; ++k) {
                float z2 = fmaxf(cf[k] + c2pv, 0.f);
                z2sh[(k + 8 * hi) * 128 + col] = z2;
            }
            __syncthreads();
            // 128-dot per row with wp: 16 threads per row, 8 cols each.
            {
                int row = tid >> 4;
                int cb  = (tid & 15) * 8;
                float p = 0.f;
#pragma unroll
                for (int c = 0; c < 8; ++c)
                    p = fmaf(z2sh[row * 128 + cb + c], wp_sh[cb + c], p);
                atomicAdd(&rowacc[row], p);
            }
            __syncthreads();
            if (tid < 16) {
                float d = rowacc[tid] + cst;
                outp[i * NN + j0 + tid] = 1.f / (1.f + expf(-d));
            }
            __syncthreads(); // before next tile overwrites z2sh/rowacc
        }
    }

    if (PASS == 0) {
        atomicAdd(&wsum[col], accS);
        atomicAdd(&wsq[col],  accQ);
        __syncthreads();
        if (tid < 128) {
            atomicAdd(&bn2sum[tid], wsum[tid]);
            atomicAdd(&bn2sq[tid],  wsq[tid]);
        }
    }
}

// ---------------------------------------------------------------------------
// Launcher.
// d_in flattening assumption: top-level dict order (x, edge_index, edge_weight,
// params) with params JAX-tree-flattened (sorted dict keys at every level):
//   3..8  : bn g0,b0,g1,b1,g2,b2
//   9..18 : head W1,W2,W3,b1,b2,b3,bb1,bb2,g1,g2
//   19+19L: layer L leaves, ASCII order:
//     +0 W_c +1 W_f +2 W_i +3 W_o +4 b_c +5 b_f +6 b_i +7 b_o
//     +8 conv_c_W +9 conv_c_b +10 conv_f_W +11 conv_f_b +12 conv_i_W
//     +13 conv_i_b +14 conv_o_W +15 conv_o_b +16 w_c_f +17 w_c_i +18 w_c_o
// ---------------------------------------------------------------------------
extern "C" void kernel_launch(void* const* d_in, const int* in_sizes, int n_in,
                              void* d_out, int out_size, void* d_ws, size_t ws_size,
                              hipStream_t stream) {
    (void)in_sizes; (void)n_in; (void)out_size; (void)ws_size;

    const float* x = (const float*)d_in[0];
    float* ws = (float*)d_ws;

    float* h      = ws + O_H;
    float* hn     = ws + O_HN;
    float* mu     = ws + O_MU;
    float* var    = ws + O_VAR;
    float* A      = ws + O_A;
    float* B      = ws + O_B;
    float* bn1sum = ws + O_ACC;
    float* bn1sq  = ws + O_ACC + 256;
    float* bn2sum = ws + O_ACC + 512;
    float* bn2sq  = ws + O_ACC + 640;
    float* scale1 = ws + O_SCALE1;
    float* shift1 = ws + O_SHIFT1;
    float* c2p    = ws + O_C2P;
    float* wp     = ws + O_WP;
    float* cst    = ws + O_CONST;
    _Float16* W2hT = (_Float16*)((char*)d_ws + O_W2HT_BYTES);
    float* outp   = (float*)d_out;

    zero_accum_kernel<<<1, 1024, 0, stream>>>(ws + O_ACC);

    // --- 3 GCLSTM layers ---
    for (int L = 0; L < 3; ++L) {
        const int base = 19 + 19 * L;
        const float* Wc  = (const float*)d_in[base + 0];
        const float* Wi  = (const float*)d_in[base + 2];
        const float* Wo  = (const float*)d_in[base + 3];
        const float* bc  = (const float*)d_in[base + 4];
        const float* bi  = (const float*)d_in[base + 6];
        const float* bo  = (const float*)d_in[base + 7];
        const float* ccb = (const float*)d_in[base + 9];
        const float* cib = (const float*)d_in[base + 13];
        const float* cob = (const float*)d_in[base + 15];
        const float* wco = (const float*)d_in[base + 18];
        const float* gbn = (const float*)d_in[3 + 2 * L];
        const float* bbn = (const float*)d_in[4 + 2 * L];

        const float* hin = (L == 0) ? x : h;
        const int fin = (L == 0) ? 64 : 128;
        gates_kernel<<<NN, 128, 0, stream>>>(hin, fin, Wi, Wc, Wo,
                                             bi, bc, bo, cib, ccb, cob, wco, hn);
        bn_stats_kernel<<<HIDD, 256, 0, stream>>>(hn, mu, var);
        bn_apply_kernel<<<NN, 128, 0, stream>>>(hn, mu, var, gbn, bbn, h, L > 0);
    }

    // --- head decomposition ---
    const float* W1  = (const float*)d_in[9];
    const float* W2  = (const float*)d_in[10];
    const float* W3  = (const float*)d_in[11];
    const float* b1  = (const float*)d_in[12];
    const float* b2  = (const float*)d_in[13];
    const float* b3  = (const float*)d_in[14];
    const float* bb1 = (const float*)d_in[15];
    const float* bb2 = (const float*)d_in[16];
    const float* g1  = (const float*)d_in[17];
    const float* g2  = (const float*)d_in[18];

    ab_kernel<<<NN, 256, 0, stream>>>(h, W1, b1, A, B);
    bn1_stats_kernel<<<NN, 256, 0, stream>>>(A, B, bn1sum, bn1sq);
    prep1_kernel<<<1, 256, 0, stream>>>(bn1sum, bn1sq, g1, bb1, b2, W2,
                                        scale1, shift1, c2p);
    w2ht_kernel<<<HIDD, 256, 0, stream>>>(scale1, W2, W2hT);

    // pass 0: BN2 stats over all 589824 pairs (WMMA)
    pair_head_kernel<0><<<NN * 3, 256, 0, stream>>>(A, B, W2hT, c2p,
                                                    bn2sum, bn2sq,
                                                    nullptr, nullptr, nullptr);
    prep2_kernel<<<1, 128, 0, stream>>>(bn2sum, bn2sq, g2, bb2, W3, b3, wp, cst);
    // pass 1: final output (WMMA)
    pair_head_kernel<1><<<NN * 3, 256, 0, stream>>>(A, B, W2hT, c2p,
                                                    bn2sum, bn2sq,
                                                    wp, cst, outp);
}
